// MEGNet_NodeConvolution_7499012898894
// MI455X (gfx1250) — matrix operand order repaired and checked by reading, not compiled, not run on hardware
//
#include <hip/hip_runtime.h>

#define N_NODES 100000
#define N_EDGES 1600000
#define DFEAT 64

typedef __bf16 bf16;
typedef __attribute__((ext_vector_type(16))) __bf16 v16bf;
typedef __attribute__((ext_vector_type(8)))  float  v8f;
typedef __attribute__((ext_vector_type(4)))  unsigned int v4u;

union ABfrag { v16bf v; v4u q[2]; };

__device__ __forceinline__ bf16 f2bf(float f) {
  unsigned u = __builtin_bit_cast(unsigned, f);
  unsigned r = u + 0x7FFFu + ((u >> 16) & 1u);   // round-to-nearest-even
  unsigned short h = (unsigned short)(r >> 16);
  return __builtin_bit_cast(bf16, h);
}

// ---------------- zero workspace ----------------
__global__ void zero_kernel(float4* __restrict__ p, int n4) {
  int i = blockIdx.x * blockDim.x + threadIdx.x;
  if (i < n4) p[i] = make_float4(0.f, 0.f, 0.f, 0.f);
}

// ---------------- edge scatter (sums + counts) ----------------
__global__ __launch_bounds__(256) void scatter_kernel(
    const int* __restrict__ eidx,     // edge_index row 0 (src), int32
    const float* __restrict__ eattr,  // [E, 64]
    float* __restrict__ sums,         // [N, 64]
    float* __restrict__ counts) {     // [N]
  int i = blockIdx.x * 256 + threadIdx.x;   // i in [0, E*64)
  int e = i >> 6;
  int c = i & 63;
  int src = eidx[e];                  // wave-uniform (64 consecutive threads share e)
  atomicAdd(&sums[src * DFEAT + c], eattr[i]);
  if (c == 0) atomicAdd(&counts[src], 1.0f);
}

// ---------------- fused GEMM + bias + ReLU + BN-stats ----------------
// Workgroup: 128 threads = 4 waves; wave w computes rows [blk*64 + 16w, +16) x 64 cols.
template <int KDIM, bool FIRST>
__global__ __launch_bounds__(128) void gemm_kernel(
    const float* __restrict__ ain,    // FIRST: x [N,64]; else h_prev [N,64]
    const float* __restrict__ sums,   // FIRST only
    const float* __restrict__ counts, // FIRST only
    const float* __restrict__ gattr,  // FIRST only [512,64]
    const int* __restrict__ batch,    // FIRST only [N], int32
    const float* __restrict__ scale,  // !FIRST: BN scale of prev layer [64]
    const float* __restrict__ shift,  // !FIRST: BN shift of prev layer [64]
    const float* __restrict__ W,      // [KDIM, 64] row-major
    const float* __restrict__ bias,   // [64]
    float* __restrict__ hout,         // [N, 64] (pre-BN, post-ReLU)
    float* __restrict__ stats) {      // [128]: col sums then col sumsq
  constexpr int KS = KDIM + 8;        // bf16 row stride (keeps rows 16B-aligned)
  __shared__ __align__(16) bf16 As[64 * KS];   // A tile, row-major
  __shared__ __align__(16) bf16 Ws[64 * KS];   // W transposed: Wt[n][k]
  __shared__ float ssum[DFEAT];
  __shared__ float ssq[DFEAT];

  const int tid = threadIdx.x;
  const int rowbase = blockIdx.x * 64;
  if (tid < DFEAT) { ssum[tid] = 0.f; ssq[tid] = 0.f; }

  // Stage transposed weights (bf16)
  for (int i = tid; i < 64 * KDIM; i += 128) {
    int n = i / KDIM, k = i - n * KDIM;
    Ws[n * KS + k] = f2bf(W[k * DFEAT + n]);
  }
  // Stage A tile (bf16), fusing scatter-mean / gather / BN-affine
  for (int i = tid; i < 64 * KDIM; i += 128) {
    int r = i / KDIM, k = i - r * KDIM;
    int row = rowbase + r;
    float v = 0.f;
    if (row < N_NODES) {
      if (FIRST) {
        if (k < 64) {
          v = ain[row * DFEAT + k];
        } else if (k < 128) {
          float cnt = counts[row];
          v = sums[row * DFEAT + (k - 64)] / fmaxf(cnt, 1.0f);
        } else {
          int b = batch[row];
          v = gattr[b * DFEAT + (k - 128)];
        }
      } else {
        v = ain[row * KDIM + k] * scale[k] + shift[k];
      }
    }
    As[r * KS + k] = f2bf(v);
  }
  __syncthreads();

  const int lane = tid & 31;
  const int wave = tid >> 5;
  const int half = lane >> 4;     // 0: lanes 0-15, 1: lanes 16-31
  const int l16  = lane & 15;

  v8f zero = {0.f, 0.f, 0.f, 0.f, 0.f, 0.f, 0.f, 0.f};
  v8f acc[4] = {zero, zero, zero, zero};

  const bf16* Arow = &As[(wave * 16 + l16) * KS];
#pragma unroll
  for (int c = 0; c < KDIM / 32; ++c) {
    // A fragment: lane<16 holds K {c32+0..7, c32+16..23}; lane>=16 holds {+8..15, +24..31}
    ABfrag a;
    int kA = c * 32 + half * 8;
    a.q[0] = *(const v4u*)(Arow + kA);
    a.q[1] = *(const v4u*)(Arow + kA + 16);
#pragma unroll
    for (int t = 0; t < 4; ++t) {
      // B fragment: lane&15 = column; lane<16 holds K {c32+0..7, c32+16..23}; lane>=16 holds {+8..15, +24..31}
      const bf16* Wcol = &Ws[(t * 16 + l16) * KS];
      ABfrag b;
      int kB = c * 32 + half * 8;
      b.q[0] = *(const v4u*)(Wcol + kB);
      b.q[1] = *(const v4u*)(Wcol + kB + 16);
      acc[t] = __builtin_amdgcn_wmma_f32_16x16x32_bf16(
          false, a.v, false, b.v, (short)0, acc[t], false, false);
    }
  }

  // bias + ReLU + store + BN stats
#pragma unroll
  for (int t = 0; t < 4; ++t) {
    int col = t * 16 + l16;
    float bv = bias[col];
    float lsum = 0.f, lsq = 0.f;
#pragma unroll
    for (int r = 0; r < 8; ++r) {
      int m = half * 8 + r;                 // C/D layout: VGPR r -> M=r (lanes 0-15) / M=8+r
      int row = rowbase + wave * 16 + m;
      float v = fmaxf(acc[t][r] + bv, 0.f);
      if (row < N_NODES) {
        hout[row * DFEAT + col] = v;
        lsum += v;
        lsq += v * v;
      }
    }
    lsum += __shfl_xor(lsum, 16, 32);       // fold the two half-waves (same col)
    lsq  += __shfl_xor(lsq, 16, 32);
    if (half == 0) {
      atomicAdd(&ssum[col], lsum);
      atomicAdd(&ssq[col], lsq);
    }
  }
  __syncthreads();
  if (tid < DFEAT) {
    atomicAdd(&stats[tid], ssum[tid]);
    atomicAdd(&stats[DFEAT + tid], ssq[tid]);
  }
}

// ---------------- fold BN stats into scale/shift ----------------
__global__ void bnparams_kernel(const float* __restrict__ stats,
                                const float* __restrict__ gamma,
                                const float* __restrict__ beta,
                                float* __restrict__ scale,
                                float* __restrict__ shift) {
  int i = threadIdx.x;  // 64 threads
  const float invN = 1.0f / (float)N_NODES;
  float mean = stats[i] * invN;
  float var = stats[DFEAT + i] * invN - mean * mean;
  float s = gamma[i] * rsqrtf(var + 1e-5f);
  scale[i] = s;
  shift[i] = beta[i] - mean * s;
}

// ---------------- final in-place BN on d_out ----------------
__global__ void finalbn_kernel(float* __restrict__ out,
                               const float* __restrict__ scale,
                               const float* __restrict__ shift) {
  int i = blockIdx.x * blockDim.x + threadIdx.x;
  if (i < N_NODES * DFEAT) {
    int c = i & 63;
    out[i] = out[i] * scale[c] + shift[c];
  }
}

extern "C" void kernel_launch(void* const* d_in, const int* in_sizes, int n_in,
                              void* d_out, int out_size, void* d_ws, size_t ws_size,
                              hipStream_t stream) {
  const float* x     = (const float*)d_in[0];
  const int*   eidx  = (const int*)d_in[1];   // row 0 = src
  const float* eattr = (const float*)d_in[2];
  const float* gattr = (const float*)d_in[3];
  const int*   batch = (const int*)d_in[4];
  const float* W1 = (const float*)d_in[5];
  const float* b1 = (const float*)d_in[6];
  const float* W2 = (const float*)d_in[7];
  const float* b2 = (const float*)d_in[8];
  const float* W3 = (const float*)d_in[9];
  const float* b3 = (const float*)d_in[10];
  const float* g1 = (const float*)d_in[11];
  const float* bt1 = (const float*)d_in[12];
  const float* g2 = (const float*)d_in[13];
  const float* bt2 = (const float*)d_in[14];
  const float* g3 = (const float*)d_in[15];
  const float* bt3 = (const float*)d_in[16];

  float* ws = (float*)d_ws;
  float* buf    = ws;                 // 6,400,000 floats: sums -> h1 -> h2 (in place)
  float* counts = ws + 6400000;       // 100,000
  float* stats  = ws + 6500000;       // 3 * 128
  float* sscale = ws + 6500384;       // 3 * 64
  float* sshift = ws + 6500576;       // 3 * 64
  float* out = (float*)d_out;

  // zero sums + counts + stats (6,500,384 floats)
  const int n4 = 6500384 / 4;
  zero_kernel<<<(n4 + 255) / 256, 256, 0, stream>>>((float4*)ws, n4);

  // scatter-add edges into sums/counts (L2-resident atomics)
  scatter_kernel<<<(N_EDGES * DFEAT) / 256, 256, 0, stream>>>(eidx, eattr, buf, counts);

  const int nblk = (N_NODES + 63) / 64;  // 1563

  // layer 1: A = [x | sums/max(cnt,1) | g[batch]] (K=192) -> buf (in place over sums)
  gemm_kernel<192, true><<<nblk, 128, 0, stream>>>(
      x, buf, counts, gattr, batch, nullptr, nullptr, W1, b1, buf, stats);
  bnparams_kernel<<<1, 64, 0, stream>>>(stats, g1, bt1, sscale, sshift);

  // layer 2: A = BN(h1) (K=64) -> buf (in place)
  gemm_kernel<64, false><<<nblk, 128, 0, stream>>>(
      buf, nullptr, nullptr, nullptr, nullptr, sscale, sshift, W2, b2, buf, stats + 128);
  bnparams_kernel<<<1, 64, 0, stream>>>(stats + 128, g2, bt2, sscale + 64, sshift + 64);

  // layer 3: A = BN(h2) (K=64) -> d_out (pre-BN)
  gemm_kernel<64, false><<<nblk, 128, 0, stream>>>(
      buf, nullptr, nullptr, nullptr, nullptr, sscale + 64, sshift + 64, W3, b3, out, stats + 256);
  bnparams_kernel<<<1, 64, 0, stream>>>(stats + 256, g3, bt3, sscale + 128, sshift + 128);

  // final BN applied in place on d_out
  finalbn_kernel<<<(N_NODES * DFEAT + 255) / 256, 256, 0, stream>>>(out, sscale + 128, sshift + 128);
}
